// MyConvLSTM_73065983640218
// MI455X (gfx1250) — matrix operand order repaired and checked by reading, not compiled, not run on hardware
//
#include <hip/hip_runtime.h>
#include <hip/hip_bf16.h>

typedef __attribute__((ext_vector_type(16))) _Float16 v16h;
typedef __attribute__((ext_vector_type(8)))  float    v8f;

// Model dims
#define NB   32          // batch
#define NP   12          // time steps
#define NQ   12          // output horizon
#define ND   64          // feature dim
#define NHW  32          // H == W
#define NN   1024        // H*W
#define NCHUNK 36        // K = 9*128 = 1152 -> 36 chunks of 32

// Workspace layout (bytes)
static constexpr size_t SE_OFF    = 0;                 // 1024*64*4   = 256 KB
static constexpr size_t TE_OFF    = 256u  << 10;       // 384*64*4    =  96 KB
static constexpr size_t BPACK_OFF = 384u  << 10;       // 36*16*32*16*2 = 576 KB
static constexpr size_t H0_OFF    = 1u    << 20;       // 32*1024*64*2 = 4 MB
static constexpr size_t H1_OFF    = 5u    << 20;       // 4 MB
static constexpr size_t C_OFF     = 9u    << 20;       // 32*1024*64*4 = 8 MB
static constexpr size_t X_OFF     = 17u   << 20;       // 32*12*1024*64*2 = 48 MB

__global__ void k_zero(unsigned int* __restrict__ p, int n) {
    int i = blockIdx.x * 256 + threadIdx.x;
    if (i < n) p[i] = 0u;
}

// se = relu(SE @ w1 + b1) @ w2 + b2   (N=1024 rows, D=64)
__global__ void k_embed_se(const float* __restrict__ SE,
                           const float* __restrict__ w1, const float* __restrict__ b1,
                           const float* __restrict__ w2, const float* __restrict__ b2,
                           float* __restrict__ se_out) {
    __shared__ float s[ND], u[ND];
    int n = blockIdx.x, d = threadIdx.x;
    s[d] = SE[n * ND + d];
    __syncthreads();
    float acc = b1[d];
    for (int k = 0; k < ND; ++k) acc += s[k] * w1[k * ND + d];
    u[d] = fmaxf(acc, 0.f);
    __syncthreads();
    float o = b2[d];
    for (int k = 0; k < ND; ++k) o += u[k] * w2[k * ND + d];
    se_out[n * ND + d] = o;
}

// te: one-hot(day)||one-hot(tod) @ w1 == w1[day] + w1[7+tod]; relu; @ w2 + b2
__global__ void k_embed_te(const int* __restrict__ TE,
                           const float* __restrict__ w1, const float* __restrict__ b1,
                           const float* __restrict__ w2, const float* __restrict__ b2,
                           float* __restrict__ te_out) {
    __shared__ float u[ND];
    int r = blockIdx.x, d = threadIdx.x;      // r = b*NP + p
    int day = TE[r * 2 + 0];
    int tod = TE[r * 2 + 1];
    float acc = w1[day * ND + d] + w1[(7 + tod) * ND + d] + b1[d];
    u[d] = fmaxf(acc, 0.f);
    __syncthreads();
    float o = b2[d];
    for (int k = 0; k < ND; ++k) o += u[k] * w2[k * ND + d];
    te_out[r * ND + d] = o;
}

// x = relu(Z*w1 + b1) @ w2 + b2 + se[n] + te[b,p]  -> f16, layout (b,p,n,64)
__global__ void k_mlp_in(const float* __restrict__ Z,
                         const float* __restrict__ w1, const float* __restrict__ b1,
                         const float* __restrict__ w2, const float* __restrict__ b2,
                         const float* __restrict__ se, const float* __restrict__ te,
                         _Float16* __restrict__ xo) {
    __shared__ float u[4][ND];
    int t = threadIdx.x, sub = t >> 6, d = t & 63;
    long r = (long)blockIdx.x * 4 + sub;      // row in [0, B*P*N)
    int n  = (int)(r & (NN - 1));
    int bp = (int)(r >> 10);                  // b*NP + p
    float z = Z[r];
    u[sub][d] = fmaxf(z * w1[d] + b1[d], 0.f);
    __syncthreads();
    float o = b2[d];
    for (int k = 0; k < ND; ++k) o += u[sub][k] * w2[k * ND + d];
    o += se[n * ND + d] + te[bp * ND + d];
    xo[r * ND + d] = (_Float16)o;
}

// Pack wx||wh into per-lane WMMA B-fragment layout:
// B element e of lane l (for chunk, ntile):
//   K = chunk*32 + (l>>4)*16 + e ; n = ntile*16 + (l&15)
//   K = tap*128 + src*64 + cin ;  src 0 -> wx, 1 -> wh  (HWIO, (3,3,64,256))
__global__ void k_pack_w(const float* __restrict__ wx, const float* __restrict__ wh,
                         _Float16* __restrict__ bp) {
    int t = blockIdx.x * 256 + threadIdx.x;   // < 36*16*32*16 = 294912
    int e     = t & 15;
    int lane  = (t >> 4)  & 31;
    int nt    = (t >> 9)  & 15;
    int chunk =  t >> 13;
    int K = chunk * 32 + (lane >> 4) * 16 + e;
    int n = nt * 16 + (lane & 15);
    int tap = K >> 7, rem = K & 127;
    const float* w = (rem < 64) ? wx : wh;
    int cin = rem & 63;
    bp[t] = (_Float16)w[(tap * 64 + cin) * 256 + n];
}

// One ConvLSTM step. Block = one full image row (32 pixels) x 256 gate ch.
// 8 waves; each wave: 4 accumulators = 2 M-segments x 2 N-tiles, so each
// B fragment is reused by two WMMAs (halves global B traffic per WMMA).
__global__ void __launch_bounds__(256) k_lstm_step(
    const _Float16* __restrict__ xall,    // (B,P,N,64) f16
    const v16h*     __restrict__ Bpack,   // packed weights
    const _Float16* __restrict__ h_in,    // (B,N,64) f16
    _Float16*       __restrict__ h_out,   // (B,N,64) f16
    float*          __restrict__ c,       // (B,N,64) f32
    int p)
{
    __shared__ _Float16 tile[3 * 34 * 128];   // rows y-1..y+1, x -1..32, 128 ch
    __shared__ float    zb[32][256];
    const unsigned int* xw = (const unsigned int*)xall;
    const unsigned int* hw = (const unsigned int*)h_in;
    unsigned int* tile32 = (unsigned int*)tile;

    int t = threadIdx.x;
    int y = blockIdx.x & 31;                  // image row
    int b = blockIdx.x >> 5;                  // batch

    // Stage halo tile: rows y-1..y+1, x = -1..32, 128 ch (x:0-63, h:64-127)
    for (int idx = t; idx < 3 * 34 * 64; idx += 256) {
        int cc2 = idx & 63;                   // uint index within 128 halfs
        int sp  = idx >> 6;                   // 0..101
        int xl  = sp % 34, r = sp / 34;
        int gy = y + r - 1, gx = xl - 1;
        unsigned int v = 0u;
        if ((unsigned)gy < 32u && (unsigned)gx < 32u) {
            int pix = gy * 32 + gx;
            if (cc2 < 32) v = xw[((size_t)(b * NP + p) * NN + pix) * 32 + cc2];
            else          v = hw[((size_t)b * NN + pix) * 32 + (cc2 - 32)];
        }
        tile32[idx] = v;
    }
    __syncthreads();

    int lane = t & 31, wave = t >> 5;
    int m  = lane & 15;                       // pixel within M-segment
    int hi = lane >> 4;
    int nt0 = wave * 2;
    v8f acc00 = {};   // seg A (x 0..15),  ntile nt0
    v8f acc01 = {};   // seg A,            ntile nt0+1
    v8f acc10 = {};   // seg B (x 16..31), ntile nt0
    v8f acc11 = {};   // seg B,            ntile nt0+1

    for (int chunk = 0; chunk < NCHUNK; ++chunk) {
        int tap = chunk >> 2, q = chunk & 3;  // q selects 32-ch slice of 128
        int dy = tap / 3, dx = tap % 3;       // 0..2
        int base0 = (dy * 34 + (m + dx))      * 64 + q * 16;   // uint idx, seg A
        int base1 = (dy * 34 + (m + 16 + dx)) * 64 + q * 16;   // seg B
        union { v16h v; unsigned int u[8]; } A0, A1;
#pragma unroll
        for (int pp = 0; pp < 8; ++pp) {
            // A k order used here, same as Bpack: lane<16: e0-15->K0-15; lane>=16: e0-15->K16-31
            int K = 2 * pp + hi * 16;
            A0.u[pp] = tile32[base0 + (K >> 1)];
            A1.u[pp] = tile32[base1 + (K >> 1)];
        }
        v16h b0 = Bpack[((size_t)chunk * 16 + nt0)     * 32 + lane];
        v16h b1 = Bpack[((size_t)chunk * 16 + nt0 + 1) * 32 + lane];
        acc00 = __builtin_amdgcn_wmma_f32_16x16x32_f16(false, A0.v, false, b0,
                                                       (short)0, acc00, false, false);
        acc10 = __builtin_amdgcn_wmma_f32_16x16x32_f16(false, A1.v, false, b0,
                                                       (short)0, acc10, false, false);
        acc01 = __builtin_amdgcn_wmma_f32_16x16x32_f16(false, A0.v, false, b1,
                                                       (short)0, acc01, false, false);
        acc11 = __builtin_amdgcn_wmma_f32_16x16x32_f16(false, A1.v, false, b1,
                                                       (short)0, acc11, false, false);
    }

    // Spill z (C/D layout: lane l, vgpr v -> M = v + 8*(l>>4), N = l&15)
#pragma unroll
    for (int v = 0; v < 8; ++v) {
        int M = v + 8 * hi, col = lane & 15;
        zb[M][nt0 * 16 + col]            = acc00[v];
        zb[M][(nt0 + 1) * 16 + col]      = acc01[v];
        zb[M + 16][nt0 * 16 + col]       = acc10[v];
        zb[M + 16][(nt0 + 1) * 16 + col] = acc11[v];
    }
    __syncthreads();

    // Gates: i,f,o = hard_sigmoid; g = tanh; c' = f*c + i*tanh(g); h = o*tanh(c')
    for (int task = t; task < 32 * ND; task += 256) {
        int mm = task >> 6, d = task & 63;    // mm = pixel x in row
        float iz = zb[mm][d];
        float fz = zb[mm][64 + d];
        float gz = zb[mm][128 + d];
        float oz = zb[mm][192 + d];
        size_t ci = ((size_t)(b * 32 + y) * 32 + mm) * ND + d;
        float cold = c[ci];
        float ig = fminf(fmaxf(0.2f * iz + 0.5f, 0.f), 1.f);
        float fg = fminf(fmaxf(0.2f * fz + 0.5f, 0.f), 1.f);
        float og = fminf(fmaxf(0.2f * oz + 0.5f, 0.f), 1.f);
        float cn = fg * cold + ig * tanhf(gz);
        c[ci] = cn;
        h_out[ci] = (_Float16)(og * tanhf(cn));
    }
}

// y = relu(h @ w1 + b1) @ w2 + b2, transposed to (B,Q,N)
__global__ void k_mlp_out(const _Float16* __restrict__ hfin,
                          const float* __restrict__ w1, const float* __restrict__ b1,
                          const float* __restrict__ w2, const float* __restrict__ b2,
                          float* __restrict__ out) {
    __shared__ float hr[ND], u[ND];
    int bn = blockIdx.x;                      // b*N + n
    int d = threadIdx.x;
    hr[d] = (float)hfin[(size_t)bn * ND + d];
    __syncthreads();
    float acc = b1[d];
    for (int k = 0; k < ND; ++k) acc += hr[k] * w1[k * ND + d];
    u[d] = fmaxf(acc, 0.f);
    __syncthreads();
    if (d < NQ) {
        float o = b2[d];
        for (int k = 0; k < ND; ++k) o += u[k] * w2[k * NQ + d];
        int b = bn >> 10, n = bn & (NN - 1);
        out[((size_t)b * NQ + d) * NN + n] = o;
    }
}

extern "C" void kernel_launch(void* const* d_in, const int* in_sizes, int n_in,
                              void* d_out, int out_size, void* d_ws, size_t ws_size,
                              hipStream_t stream) {
    (void)in_sizes; (void)n_in; (void)out_size; (void)ws_size;
    const float* Z   = (const float*)d_in[0];
    const int*   TE  = (const int*)  d_in[1];
    const float* SE  = (const float*)d_in[2];
    const float* sw1 = (const float*)d_in[3];
    const float* sb1 = (const float*)d_in[4];
    const float* sw2 = (const float*)d_in[5];
    const float* sb2 = (const float*)d_in[6];
    const float* tw1 = (const float*)d_in[7];
    const float* tb1 = (const float*)d_in[8];
    const float* tw2 = (const float*)d_in[9];
    const float* tb2 = (const float*)d_in[10];
    const float* fw1 = (const float*)d_in[11];
    const float* fb1 = (const float*)d_in[12];
    const float* fw2 = (const float*)d_in[13];
    const float* fb2 = (const float*)d_in[14];
    const float* wx  = (const float*)d_in[15];
    const float* wh  = (const float*)d_in[16];
    const float* ow1 = (const float*)d_in[17];
    const float* ob1 = (const float*)d_in[18];
    const float* ow2 = (const float*)d_in[19];
    const float* ob2 = (const float*)d_in[20];

    char* ws = (char*)d_ws;
    float*    se    = (float*)(ws + SE_OFF);
    float*    te    = (float*)(ws + TE_OFF);
    _Float16* bpack = (_Float16*)(ws + BPACK_OFF);
    float*    cbuf  = (float*)(ws + C_OFF);
    _Float16* xf16  = (_Float16*)(ws + X_OFF);

    // Re-zero initial h and c every call (deterministic).
    k_zero<<<(NB * NN * ND * 2 / 4 + 255) / 256, 256, 0, stream>>>(
        (unsigned int*)(ws + H0_OFF), NB * NN * ND * 2 / 4);
    k_zero<<<(NB * NN * ND + 255) / 256, 256, 0, stream>>>(
        (unsigned int*)(ws + C_OFF), NB * NN * ND);

    k_embed_se<<<NN, ND, 0, stream>>>(SE, sw1, sb1, sw2, sb2, se);
    k_embed_te<<<NB * NP, ND, 0, stream>>>(TE, tw1, tb1, tw2, tb2, te);
    k_mlp_in<<<NB * NP * NN / 4, 256, 0, stream>>>(Z, fw1, fb1, fw2, fb2, se, te, xf16);
    k_pack_w<<<NCHUNK * 16 * 32 * 16 / 256, 256, 0, stream>>>(wx, wh, bpack);

    for (int p = 0; p < NP; ++p) {
        const _Float16* hin = (const _Float16*)(ws + ((p & 1) ? H1_OFF : H0_OFF));
        _Float16*      hout = (_Float16*)      (ws + ((p & 1) ? H0_OFF : H1_OFF));
        k_lstm_step<<<NB * 32, 256, 0, stream>>>(
            xf16, (const v16h*)bpack, hin, hout, cbuf, p);
    }

    // NP even -> final h is in H0
    k_mlp_out<<<NB * NN, ND, 0, stream>>>((const _Float16*)(ws + H0_OFF),
                                          ow1, ob1, ow2, ob2, (float*)d_out);
}
